// GCN5_7730941133133
// MI455X (gfx1250) — matrix operand order repaired and checked by reading, not compiled, not run on hardware
//
#include <hip/hip_runtime.h>

typedef __attribute__((ext_vector_type(16))) __bf16 v16bf;
typedef __attribute__((ext_vector_type(8)))  __bf16 v8bf;
typedef __attribute__((ext_vector_type(8)))  float  v8f;
typedef __attribute__((ext_vector_type(4)))  int    v4i;

#define AS1 __attribute__((address_space(1)))
#define AS3 __attribute__((address_space(3)))

#define GEMM_BM 128
#define GEMM_BN 128
#define GEMM_BK 32
#define LDS_STRIDE 40   // bf16 elems per staged row (80 B: 16B-aligned, bank-friendly)

#if __has_builtin(__builtin_amdgcn_global_load_async_to_lds_b128)
#define HAVE_ASYNC_LDS 1
#else
#define HAVE_ASYNC_LDS 0
#endif

// ---- 16-byte global->LDS copy: async (ASYNCcnt) when available ----
__device__ __forceinline__ void cp16_g2l(const __bf16* g, __bf16* l) {
#if HAVE_ASYNC_LDS
  __builtin_amdgcn_global_load_async_to_lds_b128(
      (AS1 v4i*)g, (AS3 v4i*)l, 0, 0);
#else
  *(v8bf*)l = *(const v8bf*)g;
#endif
}

__device__ __forceinline__ void wait_async_le4() {
#if HAVE_ASYNC_LDS
#if __has_builtin(__builtin_amdgcn_s_wait_asynccnt)
  __builtin_amdgcn_s_wait_asynccnt(4);
#else
  asm volatile("s_wait_asynccnt 4" ::: "memory");
#endif
#endif
}

__device__ __forceinline__ void wait_async_le0() {
#if HAVE_ASYNC_LDS
#if __has_builtin(__builtin_amdgcn_s_wait_asynccnt)
  __builtin_amdgcn_s_wait_asynccnt(0);
#else
  asm volatile("s_wait_asynccnt 0" ::: "memory");
#endif
#endif
}

// ---------------- degree / normalization ----------------

__global__ void gcn_deg_init(float* __restrict__ deg, int N) {
  int i = blockIdx.x * blockDim.x + threadIdx.x;
  if (i < N) deg[i] = 1.0f;  // self-loop contribution
}

__global__ void gcn_deg_count(float* __restrict__ deg, const int* __restrict__ dst, int E) {
  int e = blockIdx.x * blockDim.x + threadIdx.x;
  if (e < E) atomicAdd(&deg[dst[e]], 1.0f);
}

__global__ void gcn_deg_rsqrt(float* __restrict__ deg, int N) {
  int i = blockIdx.x * blockDim.x + threadIdx.x;
  if (i < N) deg[i] = rsqrtf(deg[i]);
}

// ---------------- conversions ----------------

__global__ void gcn_f32_to_bf16(const float* __restrict__ in, __bf16* __restrict__ out,
                                size_t n) {
  size_t i = (size_t)blockIdx.x * blockDim.x + threadIdx.x;
  if (i < n) out[i] = (__bf16)in[i];
}

// W[din,dout] fp32 -> Wt[dout,din] bf16 (transpose so B tiles are K-contiguous)
__global__ void gcn_wt_bf16(const float* __restrict__ W, __bf16* __restrict__ Wt,
                            int din, int dout) {
  int idx = blockIdx.x * blockDim.x + threadIdx.x;
  if (idx >= din * dout) return;
  int k = idx / dout, n = idx % dout;
  Wt[(size_t)n * din + k] = (__bf16)W[idx];
}

// ---------------- shared WMMA pieces ----------------

// one 128x128x32 step: A frag per wave (16 rows), 8 B frags, 8 WMMAs
__device__ __forceinline__ void gemm_compute_step(const __bf16* sA, const __bf16* sB,
                                                  int waveId, int lane, v8f* acc) {
  const int arow  = (waveId << 4) + (lane & 15);
  const int khalf = lane >> 4;   // ISA 7.12.2 16-bit A/B lane->K mapping
  v16bf afrag;
  {
    v8bf alo = *(const v8bf*)&sA[arow * LDS_STRIDE + khalf * 8];
    v8bf ahi = *(const v8bf*)&sA[arow * LDS_STRIDE + 16 + khalf * 8];
#pragma unroll
    for (int i = 0; i < 8; ++i) { afrag[i] = alo[i]; afrag[i + 8] = ahi[i]; }
  }
#pragma unroll
  for (int t = 0; t < 8; ++t) {
    int bcol = t * 16 + (lane & 15);
    v16bf bfrag;
    v8bf blo = *(const v8bf*)&sB[bcol * LDS_STRIDE + khalf * 8];
    v8bf bhi = *(const v8bf*)&sB[bcol * LDS_STRIDE + 16 + khalf * 8];
#pragma unroll
    for (int i = 0; i < 8; ++i) { bfrag[i] = blo[i]; bfrag[i + 8] = bhi[i]; }
    acc[t] = __builtin_amdgcn_wmma_f32_16x16x32_bf16(
        false, afrag, false, bfrag, (short)0, acc[t], false, false);
  }
}

__device__ __forceinline__ void gemm_store(float* T, const v8f* acc, int N, int dout,
                                           int rowBase, int colBase, int waveId, int lane) {
  const int mhalf = lane >> 4;   // C/D layout: col = lane%16, row = r + 8*(lane/16)
  const int ccol  = lane & 15;
#pragma unroll
  for (int t = 0; t < 8; ++t) {
    int gcol = colBase + t * 16 + ccol;
    if (gcol >= dout) continue;
#pragma unroll
    for (int r = 0; r < 8; ++r) {
      int grow = rowBase + (waveId << 4) + (mhalf << 3) + r;
      if (grow < N) T[(size_t)grow * dout + gcol] = acc[t][r];
    }
  }
}

// ---------------- WMMA GEMM, async double-buffered (dout % 128 == 0) ----------------
// T[N,dout] = H[N,din](bf16) @ Wt[dout,din](bf16)^T ; fp32 accumulate.
// Tiles raw-copied global->LDS via GLOBAL_LOAD_ASYNC_TO_LDS_B128; rows >= N read
// in-workspace garbage that only feeds unstored output rows.

__global__ __launch_bounds__(256) void gcn_gemm_wmma_async(
    const __bf16* __restrict__ H, const __bf16* __restrict__ Wt,
    float* __restrict__ T, int N, int din, int dout) {
  __shared__ __align__(16) __bf16 sA[2][GEMM_BM * LDS_STRIDE];
  __shared__ __align__(16) __bf16 sB[2][GEMM_BN * LDS_STRIDE];

  const int tid = threadIdx.x, waveId = tid >> 5, lane = tid & 31;
  const int rowBase = blockIdx.x * GEMM_BM;
  const int colBase = blockIdx.y * GEMM_BN;

  v8f acc[8];
#pragma unroll
  for (int t = 0; t < 8; ++t)
#pragma unroll
    for (int r = 0; r < 8; ++r) acc[t][r] = 0.0f;

  // stage one 128x32 A tile + 128x32 B tile: 512 16B chunks each wave-striped,
  // 4 async instructions per wave per call
  auto stage = [&](int k0, int buf) {
#pragma unroll
    for (int i = 0; i < 2; ++i) {
      int c   = waveId * 64 + i * 32 + lane;  // 0..511
      int row = c >> 2;
      int kc  = (c & 3) * 8;
      cp16_g2l(H  + (size_t)(rowBase + row) * din + k0 + kc,
               &sA[buf][row * LDS_STRIDE + kc]);
      cp16_g2l(Wt + (size_t)(colBase + row) * din + k0 + kc,
               &sB[buf][row * LDS_STRIDE + kc]);
    }
  };

  const int nk = din / GEMM_BK;
  stage(0, 0);
  for (int kt = 0; kt < nk; ++kt) {
    int buf = kt & 1;
    __syncthreads();                       // all readers of buf^1 done
    if (kt + 1 < nk) {
      stage((kt + 1) * GEMM_BK, buf ^ 1);  // prefetch next tile
      wait_async_le4();                    // current tile's copies landed
    } else {
      wait_async_le0();
    }
    __syncthreads();                       // every wave's copies visible
    gemm_compute_step(sA[buf], sB[buf], waveId, lane, acc);
  }

  gemm_store(T, acc, N, dout, rowBase, colBase, waveId, lane);
}

// ---------------- WMMA GEMM, guarded synchronous (ragged dout, e.g. 6) ----------------

__global__ __launch_bounds__(256) void gcn_gemm_wmma_generic(
    const __bf16* __restrict__ H, const __bf16* __restrict__ Wt,
    float* __restrict__ T, int N, int din, int dout) {
  __shared__ __align__(16) __bf16 sA[GEMM_BM * LDS_STRIDE];
  __shared__ __align__(16) __bf16 sB[GEMM_BN * LDS_STRIDE];

  const int tid = threadIdx.x, waveId = tid >> 5, lane = tid & 31;
  const int rowBase = blockIdx.x * GEMM_BM;
  const int colBase = blockIdx.y * GEMM_BN;

  v8f acc[8];
#pragma unroll
  for (int t = 0; t < 8; ++t)
#pragma unroll
    for (int r = 0; r < 8; ++r) acc[t][r] = 0.0f;

  v8bf zv;
#pragma unroll
  for (int i = 0; i < 8; ++i) zv[i] = (__bf16)0.0f;

  for (int k0 = 0; k0 < din; k0 += GEMM_BK) {
    __syncthreads();
#pragma unroll
    for (int i = 0; i < 2; ++i) {
      int c   = tid + i * 256;   // 0..511
      int row = c >> 2;
      int kc  = (c & 3) * 8;
      v8bf va = zv;
      int grow = rowBase + row;
      if (grow < N) va = *(const v8bf*)(H + (size_t)grow * din + k0 + kc);
      *(v8bf*)&sA[row * LDS_STRIDE + kc] = va;
      v8bf vb = zv;
      int gcol = colBase + row;
      if (gcol < dout) vb = *(const v8bf*)(Wt + (size_t)gcol * din + k0 + kc);
      *(v8bf*)&sB[row * LDS_STRIDE + kc] = vb;
    }
    __syncthreads();
    gemm_compute_step(sA, sB, waveId, lane, acc);
  }

  gemm_store(T, acc, N, dout, rowBase, colBase, waveId, lane);
}

// ---------------- aggregation ----------------

__global__ void gcn_init_agg(const float* __restrict__ t, const float* __restrict__ dinv,
                             float* __restrict__ agg, int N, int dout) {
  size_t idx = (size_t)blockIdx.x * blockDim.x + threadIdx.x;
  size_t total = (size_t)N * dout;
  if (idx >= total) return;
  int i = (int)(idx / dout);
  float dv = dinv[i];
  agg[idx] = t[idx] * dv * dv;     // self-loop term
}

__global__ void gcn_edge_agg_vec(const float* __restrict__ t, const float* __restrict__ dinv,
                                 const int* __restrict__ src, const int* __restrict__ dst,
                                 float* __restrict__ agg, int E, int dout) {
  int e = blockIdx.x * 8 + (threadIdx.x >> 5);
  if (e >= E) return;
  int lane = threadIdx.x & 31;
  int s = src[e], d = dst[e];
  float nrm = dinv[s] * dinv[d];
  const float4* trow = (const float4*)(t + (size_t)s * dout);
  float* arow = agg + (size_t)d * dout;
  __builtin_prefetch(trow + lane, 0, 0);
  int q = dout >> 2;
  for (int c4 = lane; c4 < q; c4 += 32) {
    float4 v = trow[c4];
    int c = c4 << 2;
    atomicAdd(arow + c + 0, v.x * nrm);
    atomicAdd(arow + c + 1, v.y * nrm);
    atomicAdd(arow + c + 2, v.z * nrm);
    atomicAdd(arow + c + 3, v.w * nrm);
  }
}

__global__ void gcn_edge_agg_small(const float* __restrict__ t, const float* __restrict__ dinv,
                                   const int* __restrict__ src, const int* __restrict__ dst,
                                   float* __restrict__ agg, int E, int dout) {
  int e = blockIdx.x * blockDim.x + threadIdx.x;
  if (e >= E) return;
  int s = src[e], d = dst[e];
  float nrm = dinv[s] * dinv[d];
  const float* trow = t + (size_t)s * dout;
  float* arow = agg + (size_t)d * dout;
  for (int c = 0; c < dout; ++c) atomicAdd(arow + c, trow[c] * nrm);
}

// ---------------- bias (+relu) ----------------

// hidden layers: agg(f32) + b, relu, write bf16 activations for the next GEMM
__global__ void gcn_bias_relu_bf16(const float* __restrict__ agg, const float* __restrict__ b,
                                   __bf16* __restrict__ h, int N, int dout) {
  size_t idx = (size_t)blockIdx.x * blockDim.x + threadIdx.x;
  size_t total = (size_t)N * dout;
  if (idx >= total) return;
  int c = (int)(idx % dout);
  float v = agg[idx] + b[c];
  h[idx] = (__bf16)(v > 0.0f ? v : 0.0f);
}

__global__ void gcn_bias_out(const float* __restrict__ agg, const float* __restrict__ b,
                             float* __restrict__ out, int N, int dout) {
  size_t idx = (size_t)blockIdx.x * blockDim.x + threadIdx.x;
  size_t total = (size_t)N * dout;
  if (idx >= total) return;
  int c = (int)(idx % dout);
  out[idx] = agg[idx] + b[c];
}

// ---------------- driver ----------------

extern "C" void kernel_launch(void* const* d_in, const int* in_sizes, int n_in,
                              void* d_out, int out_size, void* d_ws, size_t ws_size,
                              hipStream_t stream) {
  static const int dims[6] = {32, 256, 1024, 1024, 256, 6};

  const float* x  = (const float*)d_in[0];
  const int*   ei = (const int*)d_in[1];
  const int N = in_sizes[0] / dims[0];
  const int E = in_sizes[1] / 2;
  const int* srcIdx = ei;
  const int* dstIdx = ei + E;

  const float *Wp[5], *bp[5];
  for (int l = 0; l < 5; ++l) {
    Wp[l] = (const float*)d_in[2 + 2 * l];
    bp[l] = (const float*)d_in[3 + 2 * l];
  }

  // workspace: T(f32 N*1024) | AGG(f32 N*1024) | dinv(f32 N) | Hbf(bf16 N*1024) | Wt(bf16 1M)
  const size_t NB = (size_t)N * 1024;
  float*  T    = (float*)d_ws;
  float*  AGG  = T + NB;
  float*  dinv = AGG + NB;
  __bf16* Hbf  = (__bf16*)(dinv + N);
  __bf16* Wt   = Hbf + NB;          // also absorbs async row-overshoot reads past Hbf

  // normalization coefficients
  gcn_deg_init <<<(N + 255) / 256, 256, 0, stream>>>(dinv, N);
  gcn_deg_count<<<(E + 255) / 256, 256, 0, stream>>>(dinv, dstIdx, E);
  gcn_deg_rsqrt<<<(N + 255) / 256, 256, 0, stream>>>(dinv, N);

  // x -> bf16 activations
  {
    size_t n0 = (size_t)N * dims[0];
    gcn_f32_to_bf16<<<(int)((n0 + 255) / 256), 256, 0, stream>>>(x, Hbf, n0);
  }

  int din = dims[0];
  for (int l = 0; l < 5; ++l) {
    int dout = dims[l + 1];

    // W -> Wt (bf16, transposed: K-contiguous rows for raw tile copies)
    gcn_wt_bf16<<<(din * dout + 255) / 256, 256, 0, stream>>>(Wp[l], Wt, din, dout);

    // T = H @ W (bf16 WMMA, fp32 accumulate)
    dim3 grid((N + GEMM_BM - 1) / GEMM_BM, (dout + GEMM_BN - 1) / GEMM_BN);
    if ((dout & 127) == 0)
      gcn_gemm_wmma_async<<<grid, 256, 0, stream>>>(Hbf, Wt, T, N, din, dout);
    else
      gcn_gemm_wmma_generic<<<grid, 256, 0, stream>>>(Hbf, Wt, T, N, din, dout);

    // aggregation: self loop + edge scatter-add (fp32)
    size_t total = (size_t)N * dout;
    int eg = (int)((total + 255) / 256);
    gcn_init_agg<<<eg, 256, 0, stream>>>(T, dinv, AGG, N, dout);
    if ((dout & 127) == 0)
      gcn_edge_agg_vec<<<(E + 7) / 8, 256, 0, stream>>>(T, dinv, srcIdx, dstIdx, AGG, E, dout);
    else
      gcn_edge_agg_small<<<(E + 255) / 256, 256, 0, stream>>>(T, dinv, srcIdx, dstIdx, AGG, E, dout);

    if (l < 4) {
      gcn_bias_relu_bf16<<<eg, 256, 0, stream>>>(AGG, bp[l], Hbf, N, dout);
      din = dout;
    } else {
      gcn_bias_out<<<eg, 256, 0, stream>>>(AGG, bp[l], (float*)d_out, N, dout);
    }
  }
}
